// GRU_89197880803524
// MI455X (gfx1250) — compile-verified, matrix operand order checked by reading, not run on hardware
//
#include <hip/hip_runtime.h>
#include <stdint.h>

#define BSZ    64
#define CIN    256
#define COUT   512
#define T_STEPS 1024
#define KTOT   (CIN + COUT)   // 768
#define NGATE  (4 * COUT)     // 2048
#define NWG    32
#define NCOL   16             // c_out columns owned per workgroup

typedef __attribute__((ext_vector_type(16))) __bf16        v16bf;
typedef __attribute__((ext_vector_type(8)))  float         v8f;
typedef __attribute__((ext_vector_type(4)))  unsigned int  v4u;

union Frag { v4u q[2]; v16bf v; };

__device__ __forceinline__ unsigned short f32_to_bf16(float f) {
    union { float f; uint32_t u; } v; v.f = f;
    uint32_t u = v.u;
    return (unsigned short)((u + 0x7FFFu + ((u >> 16) & 1u)) >> 16);
}

// ws_x[t][b][c] = bf16(x[b][c][t])  (time-major, row-contiguous in c)
__global__ void cvt_x_kernel(const float* __restrict__ x, unsigned short* __restrict__ xw) {
    size_t s = (size_t)blockIdx.x * blockDim.x + threadIdx.x;   // over B*CIN*T (src-linear)
    int t = (int)(s % T_STEPS);
    int c = (int)((s / T_STEPS) % CIN);
    int b = (int)(s / ((size_t)T_STEPS * CIN));
    xw[((size_t)t * BSZ + b) * CIN + c] = f32_to_bf16(x[s]);
}

__global__ void cvt_w_kernel(const float* __restrict__ W, unsigned short* __restrict__ Ww) {
    size_t s = (size_t)blockIdx.x * blockDim.x + threadIdx.x;   // over NGATE*KTOT
    Ww[s] = f32_to_bf16(W[s]);
}

__global__ void init_kernel(unsigned short* __restrict__ hbuf, unsigned int* __restrict__ ctr) {
    int s = blockIdx.x * blockDim.x + threadIdx.x;              // over 2*B*COUT
    hbuf[s] = 0;
    if (s == 0) *ctr = 0u;
}

__global__ __launch_bounds__(256, 1)
void lstm_persistent(const unsigned short* __restrict__ xw,    // [T][B][CIN] bf16
                     const unsigned short* __restrict__ Ww,    // [NGATE][KTOT] bf16
                     const float*          __restrict__ bias,  // [NGATE]
                     unsigned short*       __restrict__ hbuf,  // [2][B][COUT] bf16
                     unsigned int*         __restrict__ ctr,
                     float*                __restrict__ out)   // [B][COUT][T]
{
    __shared__ __align__(16) unsigned short sW[64 * KTOT];     // 96 KB: this WG's 64 gate rows
    __shared__ __align__(16) float          sG[BSZ * 64];      // 16 KB: gate pre-activations

    const int tid  = threadIdx.x;
    const int wg   = blockIdx.x;          // 0..31
    const int lane = tid & 31;
    const int wave = tid >> 5;            // 0..7
    const int mt   = wave & 3;            // M tile (batch rows mt*16..mt*16+15)
    const int np   = wave >> 2;           // 0 -> gates {0,1}, 1 -> gates {2,3}

    // ---- stage weight slab into LDS (once; reused for all 1024 steps) ----
    for (int idx = tid; idx < (64 * KTOT) / 8; idx += 256) {
        int flat = idx * 8;
        int r = flat / KTOT;              // local row 0..63 = gate*16 + n
        int k = flat % KTOT;
        int g = r >> 4, n = r & 15;
        const v4u* src = reinterpret_cast<const v4u*>(
            Ww + ((size_t)(g * COUT + wg * NCOL + n)) * KTOT + k);
        *reinterpret_cast<v4u*>(&sW[r * KTOT + k]) = *src;
    }

    // ---- per-thread elementwise ownership: col fixed, 4 batch rows ----
    const int col = tid & 15;             // local c_out column 0..15
    float bi[4];
    #pragma unroll
    for (int g = 0; g < 4; ++g) bi[g] = bias[g * COUT + wg * NCOL + col];
    float c_state[4] = {0.f, 0.f, 0.f, 0.f};

    // ---- WMMA fragment geometry (per ISA 16-bit A / B layouts) ----
    const int mrow = mt * 16 + (lane & 15);
    const int koff = (lane < 16) ? 0 : 8;
    const int rB0  = (np * 2 + 0) * 16 + (lane & 15);
    const int rB1  = (np * 2 + 1) * 16 + (lane & 15);

    // x-part GEMM for one timestep (recurrence-independent, software-pipelined)
    auto xpart = [&](int t, v8f& o0, v8f& o1) {
        const unsigned short* xrow = xw + ((size_t)t * BSZ + mrow) * CIN;
        #pragma unroll
        for (int kk = 0; kk < CIN / 32; ++kk) {
            const int k0 = kk * 32;
            Frag a, b0, b1;
            a.q[0]  = *reinterpret_cast<const v4u*>(xrow + k0 + koff);
            a.q[1]  = *reinterpret_cast<const v4u*>(xrow + k0 + koff + 16);
            b0.q[0] = *reinterpret_cast<const v4u*>(&sW[rB0 * KTOT + k0 + koff]);
            b0.q[1] = *reinterpret_cast<const v4u*>(&sW[rB0 * KTOT + k0 + koff + 16]);
            b1.q[0] = *reinterpret_cast<const v4u*>(&sW[rB1 * KTOT + k0 + koff]);
            b1.q[1] = *reinterpret_cast<const v4u*>(&sW[rB1 * KTOT + k0 + koff + 16]);
            o0 = __builtin_amdgcn_wmma_f32_16x16x32_bf16(false, a.v, false, b0.v,
                                                         (short)0, o0, false, false);
            o1 = __builtin_amdgcn_wmma_f32_16x16x32_bf16(false, a.v, false, b1.v,
                                                         (short)0, o1, false, false);
        }
    };

    __syncthreads();

    // prologue: x contribution for t = 0
    v8f pip0 = {}; v8f pip1 = {};
    xpart(0, pip0, pip1);

    for (int t = 0; t < T_STEPS; ++t) {
        const int cur = t & 1;
        const unsigned short* hcur = hbuf + (size_t)cur * BSZ * COUT;
        unsigned short*       hnxt = hbuf + (size_t)(cur ^ 1) * BSZ * COUT;

        // ---- recurrent part: acc starts from pipelined x contribution ----
        v8f acc0 = pip0; v8f acc1 = pip1;
        const unsigned short* hrow = hcur + (size_t)mrow * COUT;
        #pragma unroll
        for (int kk = 0; kk < COUT / 32; ++kk) {
            const int k0 = kk * 32;
            const int kw = CIN + k0;
            Frag a, b0, b1;
            a.q[0]  = *reinterpret_cast<const v4u*>(hrow + k0 + koff);
            a.q[1]  = *reinterpret_cast<const v4u*>(hrow + k0 + koff + 16);
            b0.q[0] = *reinterpret_cast<const v4u*>(&sW[rB0 * KTOT + kw + koff]);
            b0.q[1] = *reinterpret_cast<const v4u*>(&sW[rB0 * KTOT + kw + koff + 16]);
            b1.q[0] = *reinterpret_cast<const v4u*>(&sW[rB1 * KTOT + kw + koff]);
            b1.q[1] = *reinterpret_cast<const v4u*>(&sW[rB1 * KTOT + kw + koff + 16]);
            acc0 = __builtin_amdgcn_wmma_f32_16x16x32_bf16(false, a.v, false, b0.v,
                                                           (short)0, acc0, false, false);
            acc1 = __builtin_amdgcn_wmma_f32_16x16x32_bf16(false, a.v, false, b1.v,
                                                           (short)0, acc1, false, false);
        }

        // stage gate pre-activations (D layout: M = r + 8*(lane/16), N = lane%16)
        {
            const int m_base = mt * 16 + ((lane >> 4) ? 8 : 0);
            const int n0 = (np * 2 + 0) * 16 + (lane & 15);
            const int n1 = (np * 2 + 1) * 16 + (lane & 15);
            #pragma unroll
            for (int r = 0; r < 8; ++r) {
                sG[(m_base + r) * 64 + n0] = acc0[r];
                sG[(m_base + r) * 64 + n1] = acc1[r];
            }
        }
        __syncthreads();

        // elementwise LSTM cell: i, j, f, o at col offsets 0/16/32/48
        #pragma unroll
        for (int j = 0; j < 4; ++j) {
            const int b = (tid >> 4) + 16 * j;
            float gi = sG[b * 64 +  0 + col] + bi[0];
            float gj = sG[b * 64 + 16 + col] + bi[1];
            float gf = sG[b * 64 + 32 + col] + bi[2];
            float go = sG[b * 64 + 48 + col] + bi[3];
            float si = 1.f / (1.f + __expf(-gi));
            float sf = 1.f / (1.f + __expf(-gf));
            float so = 1.f / (1.f + __expf(-go));
            float nc = c_state[j] * sf + si * tanhf(gj);
            c_state[j] = nc;
            float nh = tanhf(nc) * so;
            hnxt[b * COUT + wg * NCOL + col] = f32_to_bf16(nh);
            out[((size_t)b * COUT + wg * NCOL + col) * T_STEPS + t] = nh;
        }

        // ---- split device-wide barrier: SIGNAL, overlap work, then WAIT ----
        __threadfence();                  // publish h slice (agent scope)
        __syncthreads();                  // all threads of this WG are done
        if (tid == 0)
            __hip_atomic_fetch_add(ctr, 1u, __ATOMIC_RELEASE, __HIP_MEMORY_SCOPE_AGENT);

        // overlap: x-part GEMM for t+1 needs no cross-WG data
        pip0 = v8f{}; pip1 = v8f{};
        if (t + 1 < T_STEPS) {
            if (t + 2 < T_STEPS)
                __builtin_prefetch(xw + ((size_t)(t + 2) * BSZ + mrow) * CIN, 0, 1);
            xpart(t + 1, pip0, pip1);
        }

        if (tid == 0) {
            const unsigned int target = (unsigned int)(t + 1) * NWG;
            while (__hip_atomic_load(ctr, __ATOMIC_ACQUIRE, __HIP_MEMORY_SCOPE_AGENT) < target) {
                __builtin_amdgcn_s_sleep(2);
            }
        }
        __syncthreads();
    }
}

extern "C" void kernel_launch(void* const* d_in, const int* in_sizes, int n_in,
                              void* d_out, int out_size, void* d_ws, size_t ws_size,
                              hipStream_t stream) {
    const float* x  = (const float*)d_in[0];
    const float* W  = (const float*)d_in[1];
    const float* b  = (const float*)d_in[2];
    float* out = (float*)d_out;

    unsigned char* ws = (unsigned char*)d_ws;
    size_t off = 0;
    unsigned short* xw = (unsigned short*)(ws + off); off += (size_t)T_STEPS * BSZ * CIN * 2;  // 32 MB
    unsigned short* Ww = (unsigned short*)(ws + off); off += (size_t)NGATE * KTOT * 2;          //  3 MB
    unsigned short* hb = (unsigned short*)(ws + off); off += (size_t)2 * BSZ * COUT * 2;        // 128 KB
    unsigned int*  ctr = (unsigned int*)(ws + off);

    cvt_x_kernel<<<(BSZ * CIN * T_STEPS) / 256, 256, 0, stream>>>(x, xw);
    cvt_w_kernel<<<((size_t)NGATE * KTOT) / 256, 256, 0, stream>>>(W, Ww);
    init_kernel<<<(2 * BSZ * COUT) / 256, 256, 0, stream>>>(hb, ctr);
    lstm_persistent<<<NWG, 256, 0, stream>>>(xw, Ww, b, hb, ctr, out);
}
